// SetAbstraction_22978075033666
// MI455X (gfx1250) — matrix operand order repaired and checked by reading, not compiled, not run on hardware
//
#include <hip/hip_runtime.h>
#include <hip/hip_bf16.h>

// ---------------------------------------------------------------------------
// PointNet++ SetAbstraction for MI455X (gfx1250, wave32, WMMA + TDM).
// Pipeline (all on `stream`; BN needs global stats -> kernel-level sync):
//   0) zero stat accumulators, convert weights to padded f16 (TDM source)
//   1) FPS            (one 1024-thread block per batch)
//   2) kNN top-32     (u64 (dist,idx) keys, stable-argsort order)
//   3) layer1: gather(points+xyz_norm) -> LDS -> v_wmma_f32_16x16x32_f16
//   4) BN1 finalize   5) layer2 (WMMA)  6) BN2 finalize
//   7) layer3 (WMMA)  8) BN3 finalize   9) BN3+ReLU+maxpool(K) -> d_out
// Activations stored pre-BN in f16, layout [B][S][P][C] (channel-contiguous)
// so GEMM epilogue stores and next-layer gathers are 16B/128B vector ops.
// W tiles are staged into LDS via the Tensor Data Mover when available.
// ---------------------------------------------------------------------------

#define Bb   16
#define Nn   4096
#define Ss   1024
#define Kk   32

typedef __attribute__((ext_vector_type(16))) _Float16 v16h;
typedef __attribute__((ext_vector_type(8)))  _Float16 v8h;
typedef __attribute__((ext_vector_type(8)))  float    v8f;

#if defined(__has_builtin)
#if __has_builtin(__builtin_amdgcn_tensor_load_to_lds) && \
    __has_builtin(__builtin_amdgcn_s_wait_tensorcnt)
#define HAVE_TDM 1
#endif
#endif

#ifdef HAVE_TDM
typedef unsigned int v4u_ __attribute__((ext_vector_type(4)));
typedef int          v8i_ __attribute__((ext_vector_type(8)));
typedef int          v4i_ __attribute__((ext_vector_type(4)));

// 2D tile load global->LDS through the Tensor Data Mover.
// D# built per CDNA5 ISA 8.3/8.4: group0 = {count=1, lds_addr, global_addr,
// type=2}; group1 = {data_size=2B, tensor_dim0/1 = cols/rows, tile = tensor,
// tensor_dim0_stride = cols}. Groups 2/3 zero (<=2D tensor).
__device__ __forceinline__ void tdm_load_tile(const _Float16* gsrc,
                                              _Float16* lds,
                                              int rows, int cols) {
  const unsigned long long ga = (unsigned long long)(uintptr_t)gsrc;
  const unsigned int ldsa = (unsigned int)(uintptr_t)lds;  // LDS offset = addr[31:0]
  v4u_ g0;
  g0[0] = 1u;                                              // count=1 (valid user D#)
  g0[1] = ldsa;                                            // lds_addr
  g0[2] = (unsigned int)(ga & 0xffffffffu);                // global_addr[31:0]
  g0[3] = (unsigned int)((ga >> 32) & 0x01ffffffu) | (2u << 30); // [56:32] | type=2
  v8i_ g1;
  g1[0] = (1 << 16);                                       // data_size=2B, wg_mask=0
  g1[1] = (cols & 0xffff) << 16;                           // tensor_dim0[15:0]
  g1[2] = ((rows & 0xffff) << 16);                         // td0 hi=0 | tensor_dim1 lo
  g1[3] = ((cols & 0xffff) << 16);                         // td1 hi=0 | tile_dim0
  g1[4] = (rows & 0xffff);                                 // tile_dim1 (tile_dim2=0)
  g1[5] = cols;                                            // tensor_dim0_stride[31:0]
  g1[6] = 0;
  g1[7] = 0;
  const v4i_ gz = {0, 0, 0, 0};
#if __clang_major__ >= 23
  const v8i_ gz8 = {0, 0, 0, 0, 0, 0, 0, 0};
  __builtin_amdgcn_tensor_load_to_lds(g0, g1, gz, gz, gz8, 0);
#else
  __builtin_amdgcn_tensor_load_to_lds(g0, g1, gz, gz, 0);
#endif
}
#endif  // HAVE_TDM

// Stage the f16 weight tile into LDS: TDM when available, vector copy else.
__device__ __forceinline__ void stage_weights(const _Float16* __restrict__ Wh,
                                              _Float16* Wl, int elems) {
#ifdef HAVE_TDM
  if (threadIdx.x < 32) {                 // wave 0 only; EXEC all-ones in-branch
    tdm_load_tile(Wh, Wl, 1, elems);      // 1 x elems row (contiguous tile)
    __builtin_amdgcn_s_wait_tensorcnt(0);
  }
#else
  for (int t = threadIdx.x * 8; t < elems; t += 256 * 8)
    *(v8h*)&Wl[t] = *(const v8h*)&Wh[t];
#endif
}

// ---------------- WMMA fragment loaders (wave32, contiguous k order in A and B) ----
// A (16x32 f16): lane l: m = l&15, g = l>>4; halves 0..15 = K[g*16 .. g*16+15]
__device__ __forceinline__ v16h load_a_frag(const _Float16* Wl, int cinp,
                                            int mt, int kb, int lane) {
  const int g = lane >> 4, m = lane & 15;
  const _Float16* p = Wl + (mt * 16 + m) * cinp + kb + g * 16;
  v16h a;
#pragma unroll
  for (int i = 0; i < 16; ++i) a[i] = p[i];
  return a;
}
// B (32x16 f16): lane l: n = l&15, g = l>>4; halves 0..15 = K[g*16 .. g*16+15]
__device__ __forceinline__ v16h load_b_frag(const _Float16* Xl, int cinp,
                                            int nt, int kb, int lane) {
  const int g = lane >> 4, n = lane & 15;
  const _Float16* p = Xl + (nt * 16 + n) * cinp + kb + g * 16;
  v16h b;
#pragma unroll
  for (int i = 0; i < 16; ++i) b[i] = p[i];
  return b;
}

// ---------------- 0a) zero BN stat block -----------------------------------
__global__ void zero_stats_kernel(float* p) { p[threadIdx.x] = 0.0f; }

// ---------------- 0b) weights -> padded f16 (TDM-friendly source) ----------
// Wh1: [64][96] (67 real cols, zero pad), Wh2: [64][64], Wh3: [128][64]
__global__ void convert_weights_kernel(const float* __restrict__ W1,
                                       const float* __restrict__ W2,
                                       const float* __restrict__ W3,
                                       _Float16* __restrict__ Wh1,
                                       _Float16* __restrict__ Wh2,
                                       _Float16* __restrict__ Wh3) {
  const int t = blockIdx.x * 256 + threadIdx.x;
  if (t < 64 * 96) {
    const int o = t / 96, c = t % 96;
    Wh1[t] = (_Float16)(c < 67 ? W1[o * 67 + c] : 0.0f);
  } else if (t < 64 * 96 + 64 * 64) {
    const int u = t - 64 * 96;
    Wh2[u] = (_Float16)W2[u];
  } else if (t < 64 * 96 + 64 * 64 + 128 * 64) {
    const int u = t - 64 * 96 - 64 * 64;
    Wh3[u] = (_Float16)W3[u];
  }
}

// ---------------- 1) farthest point sampling -------------------------------
__global__ void __launch_bounds__(1024)
fps_kernel(const float* __restrict__ xyz, float* __restrict__ newxyz,
           int* __restrict__ fps_idx, float* __restrict__ out) {
  const int b = blockIdx.x, tid = threadIdx.x;
  const int lane = tid & 31, wid = tid >> 5;
  __shared__ float rd[32];
  __shared__ int   ri[32];
  __shared__ int   farS;
  const float* xb = xyz + (size_t)b * 3 * Nn;

  float px[4], py[4], pz[4], dist[4];
#pragma unroll
  for (int j = 0; j < 4; ++j) {
    const int n = tid + j * 1024;
    px[j] = xb[n]; py[j] = xb[Nn + n]; pz[j] = xb[2 * Nn + n];
    dist[j] = 1e10f;
  }
  int far = 0;
  for (int s = 0; s < Ss; ++s) {
    const float cx = xb[far], cy = xb[Nn + far], cz = xb[2 * Nn + far];
    if (tid == 0) {
      fps_idx[b * Ss + s] = far;
      newxyz[(b * Ss + s) * 3 + 0] = cx;
      newxyz[(b * Ss + s) * 3 + 1] = cy;
      newxyz[(b * Ss + s) * 3 + 2] = cz;
      out[(b * 3 + 0) * Ss + s] = cx;       // new_xyz output [B,3,S]
      out[(b * 3 + 1) * Ss + s] = cy;
      out[(b * 3 + 2) * Ss + s] = cz;
    }
    float bd = -1.0f; int bi = 0;
#pragma unroll
    for (int j = 0; j < 4; ++j) {
      const float dx = px[j] - cx, dy = py[j] - cy, dz = pz[j] - cz;
      dist[j] = fminf(dist[j], dx * dx + dy * dy + dz * dz);
      const int n = tid + j * 1024;
      if (dist[j] > bd) { bd = dist[j]; bi = n; }
    }
    // wave32 argmax (tie -> lowest index, matches jnp.argmax)
#pragma unroll
    for (int off = 16; off; off >>= 1) {
      const float od = __shfl_xor(bd, off, 32);
      const int   oi = __shfl_xor(bi, off, 32);
      if (od > bd || (od == bd && oi < bi)) { bd = od; bi = oi; }
    }
    if (lane == 0) { rd[wid] = bd; ri[wid] = bi; }
    __syncthreads();
    if (tid < 32) {
      float d2 = rd[tid]; int i2 = ri[tid];
#pragma unroll
      for (int off = 16; off; off >>= 1) {
        const float od = __shfl_xor(d2, off, 32);
        const int   oi = __shfl_xor(i2, off, 32);
        if (od > d2 || (od == d2 && oi < i2)) { d2 = od; i2 = oi; }
      }
      if (tid == 0) farS = i2;
    }
    __syncthreads();
    far = farS;
  }
}

// ---------------- 2) kNN (top-32, stable argsort order) --------------------
__global__ void __launch_bounds__(64)
knn_kernel(const float* __restrict__ xyz, const float* __restrict__ newxyz,
           int* __restrict__ knn) {
  __shared__ float dl[2][Nn];                       // 32 KB
  const int tid = threadIdx.x, lane = tid & 31, w = tid >> 5;
  const int q = blockIdx.x * 2 + w;
  const int b = q >> 10;
  const float* xb = xyz + (size_t)b * 3 * Nn;
  const float qx = newxyz[q * 3 + 0], qy = newxyz[q * 3 + 1], qz = newxyz[q * 3 + 2];

  for (int i = 0; i < Nn / 32; ++i) {
    const int n = i * 32 + lane;
    const float dx = xb[n] - qx, dy = xb[Nn + n] - qy, dz = xb[2 * Nn + n] - qz;
    dl[w][n] = dx * dx + dy * dy + dz * dz;
  }
  __syncthreads();

  unsigned long long prev = 0ULL;
  for (int k = 0; k < Kk; ++k) {
    unsigned long long best = ~0ULL;
    for (int i = 0; i < Nn / 32; ++i) {
      const int n = lane + i * 32;
      const unsigned long long key =
          ((unsigned long long)__float_as_uint(dl[w][n]) << 32) | (unsigned)n;
      if ((k == 0 || key > prev) && key < best) best = key;
    }
#pragma unroll
    for (int off = 16; off; off >>= 1) {
      const unsigned int lo = (unsigned)__shfl_xor((int)(best & 0xffffffffu), off, 32);
      const unsigned int hi = (unsigned)__shfl_xor((int)(best >> 32), off, 32);
      const unsigned long long o = ((unsigned long long)hi << 32) | lo;
      if (o < best) best = o;
    }
    if (lane == 0) knn[q * Kk + k] = (int)(best & 0xffffffffu);
    prev = best;
  }
}

// ---------------- shared GEMM epilogue -------------------------------------
// C/D frag: element r of acc -> (o = mtBase + 8*g + r, p = ntBase + (lane&15)).
// 8 consecutive o live in 8 consecutive acc floats -> one 16B v8h store.
template <int CO, int MT>
__device__ __forceinline__ void gemm_epilogue(const v8f (&acc)[MT][2],
                                              const float* __restrict__ bias,
                                              _Float16* __restrict__ actOut,
                                              int q, int lane,
                                              float* sS, float* sQ) {
  const int g = lane >> 4, nlo = lane & 15;
#pragma unroll
  for (int mt = 0; mt < MT; ++mt)
#pragma unroll
    for (int nt = 0; nt < 2; ++nt) {
      const int p  = nt * 16 + nlo;
      const int o0 = mt * 16 + 8 * g;
      v8h hv;
#pragma unroll
      for (int r = 0; r < 8; ++r) {
        const float v = acc[mt][nt][r] + bias[o0 + r];
        atomicAdd(&sS[o0 + r], v);
        atomicAdd(&sQ[o0 + r], v * v);
        hv[r] = (_Float16)v;
      }
      *(v8h*)&actOut[((size_t)q * Kk + p) * CO + o0] = hv;   // 16B aligned
    }
}

// ---------------- 3) layer 1: gather + WMMA GEMM + BN stats ----------------
// One wave per (b,s); C[64,32] = W1[64,67pad96] * X[67pad96,32]
__global__ void __launch_bounds__(256)
layer1_kernel(const float* __restrict__ xyz, const float* __restrict__ points,
              const float* __restrict__ newxyz, const int* __restrict__ knn,
              const _Float16* __restrict__ Wh, const float* __restrict__ bias,
              _Float16* __restrict__ actOut,
              float* __restrict__ gsum, float* __restrict__ gsq) {
  constexpr int CINP = 96, CO = 64, MT = CO / 16;
  __shared__ _Float16 Wl[CO * CINP];                // 12 KB
  __shared__ _Float16 Xl[8][32 * CINP];             // 48 KB
  __shared__ float sS[CO], sQ[CO];

  const int tid = threadIdx.x, lane = tid & 31, w = tid >> 5;
  for (int c = tid; c < CO; c += 256) { sS[c] = 0.0f; sQ[c] = 0.0f; }

  __builtin_prefetch(Wh, 0, 3);                     // global_prefetch_b8
  stage_weights(Wh, Wl, CO * CINP);                 // TDM or vector copy

  const int q = blockIdx.x * 8 + w;
  const int b = q >> 10;
  const float qx = newxyz[q * 3 + 0], qy = newxyz[q * 3 + 1], qz = newxyz[q * 3 + 2];
  {
    const int p = lane;
    const int n = knn[q * Kk + p];
    _Float16* xr = &Xl[w][p * CINP];
    const float* pts = points + (size_t)b * 64 * Nn + n;
#pragma unroll 4
    for (int c = 0; c < 64; ++c) xr[c] = (_Float16)pts[(size_t)c * Nn];
    const float* xz = xyz + (size_t)b * 3 * Nn + n;
    xr[64] = (_Float16)(xz[0]      - qx);
    xr[65] = (_Float16)(xz[Nn]     - qy);
    xr[66] = (_Float16)(xz[2 * Nn] - qz);
#pragma unroll
    for (int c = 67; c < CINP; ++c) xr[c] = (_Float16)0.0f;
  }
  __syncthreads();

  v8f acc[MT][2] = {};
#pragma unroll
  for (int kb = 0; kb < CINP; kb += 32) {
    const v16h b0 = load_b_frag(&Xl[w][0], CINP, 0, kb, lane);
    const v16h b1 = load_b_frag(&Xl[w][0], CINP, 1, kb, lane);
#pragma unroll
    for (int mt = 0; mt < MT; ++mt) {
      const v16h a = load_a_frag(Wl, CINP, mt, kb, lane);
      acc[mt][0] = __builtin_amdgcn_wmma_f32_16x16x32_f16(false, a, false, b0,
                                                          (short)0, acc[mt][0], false, false);
      acc[mt][1] = __builtin_amdgcn_wmma_f32_16x16x32_f16(false, a, false, b1,
                                                          (short)0, acc[mt][1], false, false);
    }
  }
  gemm_epilogue<CO, MT>(acc, bias, actOut, q, lane, sS, sQ);
  __syncthreads();
  for (int c = tid; c < CO; c += 256) {
    atomicAdd(&gsum[c], sS[c]);
    atomicAdd(&gsq[c],  sQ[c]);
  }
}

// ---------------- BN finalize: mean/var -> scale/shift ---------------------
__global__ void finalize_stats_kernel(const float* __restrict__ sum,
                                      const float* __restrict__ sq,
                                      const float* __restrict__ gamma,
                                      const float* __restrict__ beta,
                                      float* __restrict__ scale,
                                      float* __restrict__ shift, float inv_cnt) {
  const int c = threadIdx.x;
  const float m  = sum[c] * inv_cnt;
  const float v  = sq[c] * inv_cnt - m * m;
  const float iv = rsqrtf(v + 1e-5f);
  const float sc = gamma[c] * iv;
  scale[c] = sc;
  shift[c] = beta[c] - m * sc;
}

// ---------------- layers 2/3: BN+ReLU(prev) -> WMMA GEMM + BN stats --------
template <int CO>
__global__ void __launch_bounds__(256)
layer_mid_kernel(const _Float16* __restrict__ actIn,
                 const float* __restrict__ scale, const float* __restrict__ shift,
                 const _Float16* __restrict__ Wh, const float* __restrict__ bias,
                 _Float16* __restrict__ actOut,
                 float* __restrict__ gsum, float* __restrict__ gsq) {
  constexpr int CIN = 64, MT = CO / 16;
  __shared__ _Float16 Wl[CO * CIN];                 // 8 / 16 KB
  __shared__ _Float16 Xl[8][32 * CIN];              // 32 KB
  __shared__ float sS[CO], sQ[CO];

  const int tid = threadIdx.x, lane = tid & 31, w = tid >> 5;
  for (int c = tid; c < CO; c += 256) { sS[c] = 0.0f; sQ[c] = 0.0f; }

  __builtin_prefetch(Wh, 0, 3);
  stage_weights(Wh, Wl, CO * CIN);                  // TDM or vector copy

  const int q = blockIdx.x * 8 + w;
  {
    // channel-contiguous gather: 128B per lane, coalesced across the wave
    const int p = lane;
    const _Float16* ai = actIn + ((size_t)q * Kk + p) * CIN;
    _Float16* xr = &Xl[w][p * CIN];
#pragma unroll
    for (int c = 0; c < CIN; ++c) {
      float v = (float)ai[c];
      v = fmaxf(v * scale[c] + shift[c], 0.0f);     // BN + ReLU
      xr[c] = (_Float16)v;
    }
  }
  __syncthreads();

  v8f acc[MT][2] = {};
#pragma unroll
  for (int kb = 0; kb < CIN; kb += 32) {
    const v16h b0 = load_b_frag(&Xl[w][0], CIN, 0, kb, lane);
    const v16h b1 = load_b_frag(&Xl[w][0], CIN, 1, kb, lane);
#pragma unroll
    for (int mt = 0; mt < MT; ++mt) {
      const v16h a = load_a_frag(Wl, CIN, mt, kb, lane);
      acc[mt][0] = __builtin_amdgcn_wmma_f32_16x16x32_f16(false, a, false, b0,
                                                          (short)0, acc[mt][0], false, false);
      acc[mt][1] = __builtin_amdgcn_wmma_f32_16x16x32_f16(false, a, false, b1,
                                                          (short)0, acc[mt][1], false, false);
    }
  }
  gemm_epilogue<CO, MT>(acc, bias, actOut, q, lane, sS, sQ);
  __syncthreads();
  for (int c = tid; c < CO; c += 256) {
    atomicAdd(&gsum[c], sS[c]);
    atomicAdd(&gsq[c],  sQ[c]);
  }
}

// ---------------- 9) BN3 + ReLU + max-pool over K --------------------------
// One wave per (b,s); lane handles 4 channels; reads coalesced 256B per p-row.
__global__ void __launch_bounds__(256)
pool_kernel(const _Float16* __restrict__ act, const float* __restrict__ scale,
            const float* __restrict__ shift, float* __restrict__ out) {
  const int tid = threadIdx.x, lane = tid & 31, w = tid >> 5;
  const int q = blockIdx.x * 8 + w;                 // b*S + s
  const int cbase = lane * 4;
  float sc[4], sh[4], m[4];
#pragma unroll
  for (int j = 0; j < 4; ++j) {
    sc[j] = scale[cbase + j];
    sh[j] = shift[cbase + j];
    m[j] = 0.0f;                                    // ReLU => max >= 0
  }
  const _Float16* a = act + (size_t)q * Kk * 128;
#pragma unroll 4
  for (int p = 0; p < Kk; ++p) {
    const _Float16* ap = a + p * 128 + cbase;
#pragma unroll
    for (int j = 0; j < 4; ++j)
      m[j] = fmaxf(m[j], fmaxf((float)ap[j] * sc[j] + sh[j], 0.0f));
  }
  const int b = q >> 10, s = q & (Ss - 1);
#pragma unroll
  for (int j = 0; j < 4; ++j)
    out[Bb * 3 * Ss + (size_t)(b * 128 + cbase + j) * Ss + s] = m[j];
}

// ---------------------------------------------------------------------------
extern "C" void kernel_launch(void* const* d_in, const int* in_sizes, int n_in,
                              void* d_out, int out_size, void* d_ws, size_t ws_size,
                              hipStream_t stream) {
  (void)in_sizes; (void)n_in; (void)out_size; (void)ws_size;
  const float* xyz    = (const float*)d_in[0];
  const float* points = (const float*)d_in[1];
  const float* W1  = (const float*)d_in[2];
  const float* b1  = (const float*)d_in[3];
  const float* g1  = (const float*)d_in[4];
  const float* be1 = (const float*)d_in[5];
  const float* W2  = (const float*)d_in[6];
  const float* b2  = (const float*)d_in[7];
  const float* g2  = (const float*)d_in[8];
  const float* be2 = (const float*)d_in[9];
  const float* W3  = (const float*)d_in[10];
  const float* b3  = (const float*)d_in[11];
  const float* g3  = (const float*)d_in[12];
  const float* be3 = (const float*)d_in[13];
  float* out = (float*)d_out;

  char* ws = (char*)d_ws;
  size_t off = 0;
  auto take = [&](size_t bytes) -> char* {
    char* p = ws + off;
    off += (bytes + 255) & ~(size_t)255;
    return p;
  };
  int*   fps_idx = (int*)take((size_t)Bb * Ss * 4);
  float* newxyz  = (float*)take((size_t)Bb * Ss * 3 * 4);
  int*   knn     = (int*)take((size_t)Bb * Ss * Kk * 4);
  float* stats   = (float*)take(1024 * 4);
  float* sum1 = stats +   0; float* sq1 = stats +  64;
  float* sum2 = stats + 128; float* sq2 = stats + 192;
  float* sum3 = stats + 256; float* sq3 = stats + 384;
  float* sc1  = stats + 512; float* sh1 = stats + 576;
  float* sc2  = stats + 640; float* sh2 = stats + 704;
  float* sc3  = stats + 768; float* sh3 = stats + 896;
  _Float16* Wh1 = (_Float16*)take((size_t)64 * 96 * 2);
  _Float16* Wh2 = (_Float16*)take((size_t)64 * 64 * 2);
  _Float16* Wh3 = (_Float16*)take((size_t)128 * 64 * 2);
  _Float16* act1 = (_Float16*)take((size_t)Bb * Ss * Kk * 64  * 2);  //  64 MB
  _Float16* act2 = (_Float16*)take((size_t)Bb * Ss * Kk * 64  * 2);  //  64 MB
  _Float16* act3 = (_Float16*)take((size_t)Bb * Ss * Kk * 128 * 2);  // 128 MB

  const float inv_cnt = 1.0f / (float)(Bb * Kk * Ss);                // 1/524288

  zero_stats_kernel<<<1, 1024, 0, stream>>>(stats);
  convert_weights_kernel<<<72, 256, 0, stream>>>(W1, W2, W3, Wh1, Wh2, Wh3);
  fps_kernel<<<Bb, 1024, 0, stream>>>(xyz, newxyz, fps_idx, out);
  knn_kernel<<<(Bb * Ss) / 2, 64, 0, stream>>>(xyz, newxyz, knn);

  layer1_kernel<<<(Bb * Ss) / 8, 256, 0, stream>>>(xyz, points, newxyz, knn,
                                                   Wh1, b1, act1, sum1, sq1);
  finalize_stats_kernel<<<1, 64, 0, stream>>>(sum1, sq1, g1, be1, sc1, sh1, inv_cnt);

  layer_mid_kernel<64><<<(Bb * Ss) / 8, 256, 0, stream>>>(act1, sc1, sh1, Wh2, b2,
                                                          act2, sum2, sq2);
  finalize_stats_kernel<<<1, 64, 0, stream>>>(sum2, sq2, g2, be2, sc2, sh2, inv_cnt);

  layer_mid_kernel<128><<<(Bb * Ss) / 8, 256, 0, stream>>>(act2, sc2, sh2, Wh3, b3,
                                                           act3, sum3, sq3);
  finalize_stats_kernel<<<1, 128, 0, stream>>>(sum3, sq3, g3, be3, sc3, sh3, inv_cnt);

  pool_kernel<<<(Bb * Ss) / 8, 256, 0, stream>>>(act3, sc3, sh3, out);
}
